// MoEFluxAttnProcessor2_0_50878182588813
// MI455X (gfx1250) — compile-verified, matrix operand order checked by reading, not run on hardware
//
#include <hip/hip_runtime.h>
#include <hip/hip_bf16.h>
#include <stdint.h>

// Problem constants (B=1)
#define SLEN 2048
#define DIM  3072
#define NH   24
#define HDIM 128
#define NE   4
#define NR   8

typedef __attribute__((ext_vector_type(16))) __bf16 v16bf;
typedef __attribute__((ext_vector_type(8)))  float  v8f;

typedef unsigned int tdm_u32x4 __attribute__((ext_vector_type(4)));
typedef int          tdm_i32x8 __attribute__((ext_vector_type(8)));
typedef int          tdm_i32x4 __attribute__((ext_vector_type(4)));

#if defined(__HIP_DEVICE_COMPILE__) && __has_builtin(__builtin_amdgcn_tensor_load_to_lds)
#define USE_TDM 1
#else
#define USE_TDM 0
#endif

__device__ __forceinline__ unsigned short f2bf(float f) {
  unsigned int u = __float_as_uint(f);
  unsigned int r = (u + 0x7FFFu + ((u >> 16) & 1u)) >> 16;  // RNE
  return (unsigned short)r;
}

// Load a 16-element bf16 fragment (two 16B LDS reads) — caller resolves the
// per-lane (row, k-chunk) mapping per the CDNA5 16-bit A/B VGPR layout.
__device__ __forceinline__ v16bf ld_frag(const unsigned short* p) {
  union { v16bf v; uint4 q[2]; } u;
  u.q[0] = *(const uint4*)(p);
  u.q[1] = *(const uint4*)(p + 16);
  return u.v;
}

__device__ __forceinline__ v8f wmma_bf16(v16bf a, v16bf b, v8f c) {
  return __builtin_amdgcn_wmma_f32_16x16x32_bf16(false, a, false, b, (short)0, c,
                                                 false, false);
}

__device__ __forceinline__ v8f zero8() {
  v8f z;
#pragma unroll
  for (int i = 0; i < 8; ++i) z[i] = 0.f;
  return z;
}

#if USE_TDM
// TDM: DMA a [rows x 128] bf16 tile (row-major, source row stride 128 elems)
// into LDS with 16 bytes of padding appended per 256B row (-> 136-ushort rows).
// pad_interval code 5 = 64 DWORDs (one 128-elem bf16 row); pad_amount code 3 =
// 4 DWORDs (16B). Descriptor layout per CDNA5 ISA §8 (D# groups 0/1).
// This toolchain uses the 6-arg builtin: (g0 u32x4, g1 i32x8, g2 i32x4,
// g3 i32x4, g4 i32x8, cpol i32).
__device__ __forceinline__ void tdm_load_tile_128(unsigned lds_addr,
                                                  const unsigned short* gptr,
                                                  int rows) {
  unsigned long long ga = (unsigned long long)(uintptr_t)gptr;
  tdm_u32x4 g0;
  g0[0] = 1u;                                   // count=1, user descriptor
  g0[1] = lds_addr;                             // lds_addr [63:32]
  g0[2] = (unsigned)ga;                         // global_addr low
  g0[3] = (unsigned)((ga >> 32) & 0x01FFFFFFu)  // global_addr [56:32]
          | (2u << 30);                         // type=2 ("image")
  tdm_i32x8 g1;
  g1[0] = (1 << 16)        // data_size = 2 bytes
        | (1 << 20)        // pad_enable
        | (5 << 22)        // pad_interval = 64 DWORDs
        | (3 << 25);       // pad_amount   = 4 DWORDs
  g1[1] = (int)(((unsigned)HDIM & 0xFFFFu) << 16);   // tensor_dim0 lo16
  g1[2] = (int)(((unsigned)SLEN & 0xFFFFu) << 16);   // tensor_dim1 lo16
  g1[3] = (int)(((unsigned)HDIM & 0xFFFFu) << 16);   // tile_dim0
  g1[4] = rows;                                      // tile_dim1
  g1[5] = HDIM;                                      // tensor_dim0_stride lo32
  g1[6] = 0;
  g1[7] = 0;
  tdm_i32x4 z4 = (tdm_i32x4)0;
  tdm_i32x8 z8 = (tdm_i32x8)0;
  __builtin_amdgcn_tensor_load_to_lds(g0, g1, z4, z4, z8, 0);
}
#endif

// ---------------- conversions ----------------

__global__ __launch_bounds__(256) void cvt_bf16_kernel(
    const float* __restrict__ in, unsigned short* __restrict__ out, size_t n) {
  size_t i = (size_t)blockIdx.x * 256 + threadIdx.x;
  if (i < n) out[i] = f2bf(in[i]);
}

// W[d][f] (fp32) -> Wt[f][d] (bf16), tiled 32x32 transpose through LDS
__global__ __launch_bounds__(256) void transpose_cvt_kernel(
    const float* __restrict__ W, unsigned short* __restrict__ Wt) {
  __shared__ float tile[32][33];
  const int tx = threadIdx.x & 31;
  const int ty = threadIdx.x >> 5;  // 0..7
  const int bx = blockIdx.x * 32;   // f
  const int by = blockIdx.y * 32;   // d
#pragma unroll
  for (int i = ty; i < 32; i += 8)
    tile[i][tx] = W[(size_t)(by + i) * DIM + bx + tx];
  __syncthreads();
#pragma unroll
  for (int i = ty; i < 32; i += 8)
    Wt[(size_t)(bx + i) * DIM + by + tx] = f2bf(tile[tx][i]);
}

// Y[s][h*128+hd] (fp32) -> Out[h][s][hd] (bf16)
__global__ __launch_bounds__(256) void split_heads_kernel(
    const float* __restrict__ Y, unsigned short* __restrict__ Out) {
  size_t i = (size_t)blockIdx.x * 256 + threadIdx.x;
  if (i >= (size_t)SLEN * DIM) return;
  int s = (int)(i / DIM);
  int c = (int)(i % DIM);
  int h = c >> 7;
  int hd = c & 127;
  Out[((size_t)h * SLEN + s) * HDIM + hd] = f2bf(Y[i]);
}

// ---------------- gating ----------------

// scores[s][e] = x[s,:].gW[:,e] + gb[e]  (one block per token)
__global__ __launch_bounds__(256) void gate_scores_kernel(
    const float* __restrict__ x, const float* __restrict__ gW,
    const float* __restrict__ gb, float* __restrict__ scores) {
  __shared__ float red[4 * 256];
  const int s = blockIdx.x;
  const int tid = threadIdx.x;
  float p0 = 0.f, p1 = 0.f, p2 = 0.f, p3 = 0.f;
  const float* xr = x + (size_t)s * DIM;
  for (int d = tid; d < DIM; d += 256) {
    float xv = xr[d];
    float4 w = *(const float4*)(gW + (size_t)d * 4);
    p0 += xv * w.x; p1 += xv * w.y; p2 += xv * w.z; p3 += xv * w.w;
  }
  red[0 * 256 + tid] = p0; red[1 * 256 + tid] = p1;
  red[2 * 256 + tid] = p2; red[3 * 256 + tid] = p3;
  __syncthreads();
  for (int off = 128; off > 0; off >>= 1) {
    if (tid < off) {
#pragma unroll
      for (int e = 0; e < 4; ++e)
        red[e * 256 + tid] += red[e * 256 + tid + off];
    }
    __syncthreads();
  }
  if (tid < 4) scores[(size_t)s * NE + tid] = red[tid * 256] + gb[tid];
}

// softmax over the SEQUENCE dim per expert column, then top-2 over experts
__global__ __launch_bounds__(256) void gate_topk_kernel(
    const float* __restrict__ scores, float* __restrict__ gates,
    int* __restrict__ idx) {
  __shared__ float buf[256];
  __shared__ float cmax[NE], csum[NE];
  const int tid = threadIdx.x;
  for (int e = 0; e < NE; ++e) {
    float mx = -3.0e38f;
    for (int s = tid; s < SLEN; s += 256) mx = fmaxf(mx, scores[(size_t)s * NE + e]);
    buf[tid] = mx;
    __syncthreads();
    for (int off = 128; off > 0; off >>= 1) {
      if (tid < off) buf[tid] = fmaxf(buf[tid], buf[tid + off]);
      __syncthreads();
    }
    float gmax = buf[0];
    __syncthreads();
    float sm = 0.f;
    for (int s = tid; s < SLEN; s += 256) sm += __expf(scores[(size_t)s * NE + e] - gmax);
    buf[tid] = sm;
    __syncthreads();
    for (int off = 128; off > 0; off >>= 1) {
      if (tid < off) buf[tid] += buf[tid + off];
      __syncthreads();
    }
    if (tid == 0) { cmax[e] = gmax; csum[e] = buf[0]; }
    __syncthreads();
  }
  for (int s = tid; s < SLEN; s += 256) {
    float p[NE];
#pragma unroll
    for (int e = 0; e < NE; ++e)
      p[e] = __expf(scores[(size_t)s * NE + e] - cmax[e]) / csum[e];
    int i0 = 0; float v0 = p[0];
#pragma unroll
    for (int e = 1; e < NE; ++e) if (p[e] > v0) { v0 = p[e]; i0 = e; }
    int i1 = (i0 == 0) ? 1 : 0; float v1 = p[i1];
#pragma unroll
    for (int e = 0; e < NE; ++e)
      if (e != i0 && p[e] > v1) { v1 = p[e]; i1 = e; }
    gates[(size_t)s * 2 + 0] = v0; gates[(size_t)s * 2 + 1] = v1;
    idx[(size_t)s * 2 + 0] = i0;   idx[(size_t)s * 2 + 1] = i1;
  }
}

// ---------------- LoRA ----------------

// down[s][which*32 + e*8 + r] = x[s,:].A[e,:,r]   (which: 0=q,1=k,2=v)
__global__ __launch_bounds__(128) void lora_down_kernel(
    const float* __restrict__ x, const float* __restrict__ qA,
    const float* __restrict__ kA, const float* __restrict__ vA,
    float* __restrict__ down) {
  __shared__ float xs[DIM];
  const int s = blockIdx.x;
  const int tid = threadIdx.x;
  const float* xr = x + (size_t)s * DIM;
  for (int d = tid; d < DIM; d += 128) xs[d] = xr[d];
  __syncthreads();
  if (tid < 96) {
    int which = tid >> 5;
    int er = tid & 31;
    int e = er >> 3, r = er & 7;
    const float* A = (which == 0 ? qA : which == 1 ? kA : vA);
    const float* ap = A + ((size_t)e * DIM) * NR + r;
    float acc = 0.f;
    for (int d = 0; d < DIM; ++d) acc += xs[d] * ap[(size_t)d * NR];
    down[(size_t)s * 96 + tid] = acc;
  }
}

// Y{q,k,v}[s][f] += sum_k gate_k * (down[s, e_k, :] . B[e_k, :, f])
__global__ __launch_bounds__(256) void lora_up_kernel(
    const float* __restrict__ down, const float* __restrict__ gates,
    const int* __restrict__ idx, const float* __restrict__ qB,
    const float* __restrict__ kB, const float* __restrict__ vB,
    float* __restrict__ Yq, float* __restrict__ Yk, float* __restrict__ Yv) {
  __shared__ float dsm[96];
  __shared__ float g[2];
  __shared__ int ix[2];
  const int s = blockIdx.x;
  const int tid = threadIdx.x;
  if (tid < 96) dsm[tid] = down[(size_t)s * 96 + tid];
  if (tid < 2) { g[tid] = gates[(size_t)s * 2 + tid]; ix[tid] = idx[(size_t)s * 2 + tid]; }
  __syncthreads();
  for (int f = tid; f < DIM; f += 256) {
    float aq = 0.f, ak = 0.f, av = 0.f;
#pragma unroll
    for (int k = 0; k < 2; ++k) {
      int e = ix[k];
      float gv = g[k];
      const float* bq = qB + ((size_t)e * NR) * DIM + f;
      const float* bk = kB + ((size_t)e * NR) * DIM + f;
      const float* bv = vB + ((size_t)e * NR) * DIM + f;
      float sq = 0.f, sk = 0.f, sv = 0.f;
#pragma unroll
      for (int r = 0; r < NR; ++r) {
        sq += dsm[0 + e * 8 + r]  * bq[(size_t)r * DIM];
        sk += dsm[32 + e * 8 + r] * bk[(size_t)r * DIM];
        sv += dsm[64 + e * 8 + r] * bv[(size_t)r * DIM];
      }
      aq += gv * sq; ak += gv * sk; av += gv * sv;
    }
    Yq[(size_t)s * DIM + f] += aq;
    Yk[(size_t)s * DIM + f] += ak;
    Yv[(size_t)s * DIM + f] += av;
  }
}

// ------- WMMA GEMM (double-buffered, software-pipelined): Y = Xb.Wt^T + bias -------

#define BM 128
#define BN 128
#define BK 32
#define LSTRIDE 40  // 32 + 8 pad (ushorts); keeps 16B alignment, avoids bank conflicts

__global__ __launch_bounds__(256) void gemm_bf16_kernel(
    const unsigned short* __restrict__ Xb,   // [SLEN][DIM] bf16
    const unsigned short* __restrict__ Wt,   // [DIM(out f)][DIM(in d)] bf16
    const float* __restrict__ bias, float* __restrict__ Y) {
  __shared__ alignas(16) unsigned short As[2][BM * LSTRIDE];
  __shared__ alignas(16) unsigned short Bs[2][BN * LSTRIDE];
  const int tid = threadIdx.x;
  const int wave = tid >> 5;
  const int lane = tid & 31;
  const int m0 = blockIdx.y * BM;
  const int n0 = blockIdx.x * BN;
  const int wm = (wave >> 1) * 32;  // 4 waves down
  const int wn = (wave & 1) * 64;   // 2 waves across
  const int lrow = tid >> 1;
  const int lcol = (tid & 1) * 16;
  const int nl = lane & 15;
  const int kb = (lane >> 4) * 8;   // per-lane k chunk base (ISA 16-bit layout)
  const int soff = lrow * LSTRIDE + lcol;

  v8f acc[2][4];
#pragma unroll
  for (int i = 0; i < 2; ++i)
#pragma unroll
    for (int j = 0; j < 4; ++j) acc[i][j] = zero8();

  const unsigned short* gA = Xb + (size_t)(m0 + lrow) * DIM + lcol;
  const unsigned short* gB = Wt + (size_t)(n0 + lrow) * DIM + lcol;

  // prologue: stage tile 0 into buffer 0
  {
    uint4 a0 = *(const uint4*)(gA);
    uint4 a1 = *(const uint4*)(gA + 8);
    uint4 b0 = *(const uint4*)(gB);
    uint4 b1 = *(const uint4*)(gB + 8);
    *(uint4*)(As[0] + soff) = a0; *(uint4*)(As[0] + soff + 8) = a1;
    *(uint4*)(Bs[0] + soff) = b0; *(uint4*)(Bs[0] + soff + 8) = b1;
  }
  __syncthreads();

  int buf = 0;
  for (int k0 = 0; k0 < DIM; k0 += BK) {
    const bool hasNext = (k0 + BK) < DIM;
    uint4 ra0, ra1, rb0, rb1;
    if (hasNext) {
      // issue next tile's global loads; they fly while WMMAs execute below
      ra0 = *(const uint4*)(gA + k0 + BK);
      ra1 = *(const uint4*)(gA + k0 + BK + 8);
      rb0 = *(const uint4*)(gB + k0 + BK);
      rb1 = *(const uint4*)(gB + k0 + BK + 8);
      __builtin_prefetch(gA + k0 + 2 * BK, 0, 1);
      __builtin_prefetch(gB + k0 + 2 * BK, 0, 1);
    }
    const unsigned short* Ab = As[buf];
    const unsigned short* Bb = Bs[buf];
    // batch all 6 fragment loads so ds_loads coalesce under one wait
    v16bf a0 = ld_frag(Ab + (wm + nl) * LSTRIDE + kb);
    v16bf a1 = ld_frag(Ab + (wm + 16 + nl) * LSTRIDE + kb);
    v16bf b0 = ld_frag(Bb + (wn + 0 + nl) * LSTRIDE + kb);
    v16bf b1 = ld_frag(Bb + (wn + 16 + nl) * LSTRIDE + kb);
    v16bf b2 = ld_frag(Bb + (wn + 32 + nl) * LSTRIDE + kb);
    v16bf b3 = ld_frag(Bb + (wn + 48 + nl) * LSTRIDE + kb);
    acc[0][0] = wmma_bf16(a0, b0, acc[0][0]);
    acc[1][0] = wmma_bf16(a1, b0, acc[1][0]);
    acc[0][1] = wmma_bf16(a0, b1, acc[0][1]);
    acc[1][1] = wmma_bf16(a1, b1, acc[1][1]);
    acc[0][2] = wmma_bf16(a0, b2, acc[0][2]);
    acc[1][2] = wmma_bf16(a1, b2, acc[1][2]);
    acc[0][3] = wmma_bf16(a0, b3, acc[0][3]);
    acc[1][3] = wmma_bf16(a1, b3, acc[1][3]);
    if (hasNext) {
      unsigned short* An = As[buf ^ 1];
      unsigned short* Bn = Bs[buf ^ 1];
      *(uint4*)(An + soff) = ra0; *(uint4*)(An + soff + 8) = ra1;
      *(uint4*)(Bn + soff) = rb0; *(uint4*)(Bn + soff + 8) = rb1;
    }
    __syncthreads();
    buf ^= 1;
  }

  // C/D layout: VGPR r -> M = (lane>=16 ? 8 : 0) + r, N = lane&15
  const int rl = (lane >> 4) * 8;
#pragma unroll
  for (int i = 0; i < 2; ++i)
#pragma unroll
    for (int j = 0; j < 4; ++j) {
      const int col = n0 + wn + j * 16 + nl;
      const float bcol = bias[col];
      float* yp = Y + (size_t)(m0 + wm + i * 16 + rl) * DIM + col;
#pragma unroll
      for (int r = 0; r < 8; ++r) yp[(size_t)r * DIM] = acc[i][j][r] + bcol;
    }
}

// ---------------- flash attention (bf16 WMMA, online softmax, TDM tile fill) ----------------

__global__ __launch_bounds__(128) void flash_attn_kernel(
    const unsigned short* __restrict__ Qb,  // [NH][SLEN][HDIM] bf16
    const unsigned short* __restrict__ Kb,
    const unsigned short* __restrict__ Vb,
    float* __restrict__ out) {              // [SLEN][DIM] fp32
  __shared__ alignas(16) unsigned short Qs[64 * 136];
  __shared__ alignas(16) unsigned short Ks[32 * 136];
  __shared__ alignas(16) unsigned short Vt[128 * 40];  // [hd][key]
  __shared__ alignas(16) unsigned short Ps[4 * 16 * 40];

  const int tid = threadIdx.x;
  const int wave = tid >> 5;  // 4 waves, each owns 16 q rows
  const int lane = tid & 31;
  const int nl = lane & 15;
  const int kb = (lane >> 4) * 8;
  const int h = blockIdx.y;
  const int q0 = blockIdx.x * 64;

  // stage Q tile (64 x 128) -> padded 136-ushort rows
#if USE_TDM
  if (wave == 0) {
    tdm_load_tile_128((unsigned)(uintptr_t)&Qs[0],
                      Qb + ((size_t)h * SLEN + q0) * HDIM, 64);
    __builtin_amdgcn_s_wait_tensorcnt(0);
  }
#else
  {
    const int row = tid >> 1;
    const int seg = (tid & 1) * 64;
    const unsigned short* gq = Qb + ((size_t)h * SLEN + q0 + row) * HDIM + seg;
#pragma unroll
    for (int u = 0; u < 8; ++u)
      *(uint4*)(Qs + row * 136 + seg + u * 8) = *(const uint4*)(gq + u * 8);
  }
#endif

  float m_r[8], l_r[8], alpha[8];
  v8f acc[8];
#pragma unroll
  for (int r = 0; r < 8; ++r) { m_r[r] = -3.0e38f; l_r[r] = 0.f; }
#pragma unroll
  for (int j = 0; j < 8; ++j) acc[j] = zero8();

  const float scale = 0.088388347648318447f;  // 1/sqrt(128)
  const int krow = tid >> 2;
  const int kseg = (tid & 3) * 32;
  unsigned short* pw = Ps + wave * 16 * 40;
  const int qrow0 = wave * 16;

  for (int kv0 = 0; kv0 < SLEN; kv0 += 32) {
    __syncthreads();  // prior iteration done reading Ks/Vt/Ps
    // K tile (32 x 128) via Tensor Data Mover (one wave issues the DMA)
#if USE_TDM
    if (wave == 0) {
      tdm_load_tile_128((unsigned)(uintptr_t)&Ks[0],
                        Kb + ((size_t)h * SLEN + kv0) * HDIM, 32);
    }
#else
    {
      const unsigned short* gk = Kb + ((size_t)h * SLEN + kv0 + krow) * HDIM + kseg;
#pragma unroll
      for (int u = 0; u < 4; ++u)
        *(uint4*)(Ks + krow * 136 + kseg + u * 8) = *(const uint4*)(gk + u * 8);
    }
#endif
    {  // V tile transposed into [hd][key]
      const unsigned short* gv = Vb + ((size_t)h * SLEN + kv0 + krow) * HDIM + kseg;
#pragma unroll
      for (int u = 0; u < 4; ++u) {
        union { uint4 q; unsigned short hh[8]; } t;
        t.q = *(const uint4*)(gv + u * 8);
#pragma unroll
        for (int j = 0; j < 8; ++j)
          Vt[(kseg + u * 8 + j) * 40 + krow] = t.hh[j];
      }
    }
#if USE_TDM
    if (wave == 0) __builtin_amdgcn_s_wait_tensorcnt(0);
#endif
    __syncthreads();

    // S = Q(16x128) @ K^T: two 16x16 score tiles (keys 0-15 / 16-31)
    v8f sc0 = zero8(), sc1 = zero8();
#pragma unroll
    for (int kk = 0; kk < HDIM; kk += 32) {
      v16bf a  = ld_frag(Qs + (qrow0 + nl) * 136 + kk + kb);
      v16bf b0 = ld_frag(Ks + nl * 136 + kk + kb);          // K row-major == B^T
      v16bf b1 = ld_frag(Ks + (16 + nl) * 136 + kk + kb);
      sc0 = wmma_bf16(a, b0, sc0);
      sc1 = wmma_bf16(a, b1, sc1);
    }

    // online softmax; each lane-half holds 8 rows, columns across 16 lanes
#pragma unroll
    for (int r = 0; r < 8; ++r) {
      float s0 = sc0[r] * scale;
      float s1 = sc1[r] * scale;
      float mx = fmaxf(s0, s1);
#pragma unroll
      for (int msk = 1; msk < 16; msk <<= 1)
        mx = fmaxf(mx, __shfl_xor(mx, msk, 32));
      float mnew = fmaxf(m_r[r], mx);
      float a = __expf(m_r[r] - mnew);
      float p0 = __expf(s0 - mnew);
      float p1 = __expf(s1 - mnew);
      float rs = p0 + p1;
#pragma unroll
      for (int msk = 1; msk < 16; msk <<= 1)
        rs += __shfl_xor(rs, msk, 32);
      m_r[r] = mnew;
      l_r[r] = l_r[r] * a + rs;
      alpha[r] = a;
      sc0[r] = p0;
      sc1[r] = p1;
    }
#pragma unroll
    for (int j = 0; j < 8; ++j)
#pragma unroll
      for (int r = 0; r < 8; ++r) acc[j][r] *= alpha[r];

    // stage P (16x32 bf16) into LDS in A-fragment-friendly row-major layout
    const int mrow = (lane >> 4) * 8;
#pragma unroll
    for (int r = 0; r < 8; ++r) {
      pw[(mrow + r) * 40 + nl]      = f2bf(sc0[r]);
      pw[(mrow + r) * 40 + 16 + nl] = f2bf(sc1[r]);
    }
    __syncthreads();

    // O += P(16x32) @ V(32x128): 8 output col-tiles
    v16bf ap = ld_frag(pw + nl * 40 + kb);
#pragma unroll
    for (int j = 0; j < 8; ++j) {
      v16bf bvf = ld_frag(Vt + (j * 16 + nl) * 40 + kb);
      acc[j] = wmma_bf16(ap, bvf, acc[j]);
    }
  }

  const int rl = (lane >> 4) * 8;
#pragma unroll
  for (int j = 0; j < 8; ++j) {
#pragma unroll
    for (int r = 0; r < 8; ++r) {
      int s = q0 + qrow0 + rl + r;
      int col = h * HDIM + j * 16 + nl;
      out[(size_t)s * DIM + col] = acc[j][r] / l_r[r];
    }
  }
}

// ---------------- launch ----------------

extern "C" void kernel_launch(void* const* d_in, const int* in_sizes, int n_in,
                              void* d_out, int out_size, void* d_ws, size_t ws_size,
                              hipStream_t stream) {
  (void)in_sizes; (void)n_in; (void)out_size; (void)ws_size;
  const float* x  = (const float*)d_in[0];
  const float* Wq = (const float*)d_in[1];
  const float* bq = (const float*)d_in[2];
  const float* Wk = (const float*)d_in[3];
  const float* bk = (const float*)d_in[4];
  const float* Wv = (const float*)d_in[5];
  const float* bv = (const float*)d_in[6];
  const float* gW = (const float*)d_in[7];
  const float* gb = (const float*)d_in[8];
  const float* qA = (const float*)d_in[9];
  const float* qB = (const float*)d_in[10];
  const float* kA = (const float*)d_in[11];
  const float* kB = (const float*)d_in[12];
  const float* vA = (const float*)d_in[13];
  const float* vB = (const float*)d_in[14];
  float* out = (float*)d_out;

  char* ws = (char*)d_ws;
  size_t off = 0;
  auto alloc = [&](size_t bytes) {
    size_t o = off;
    off += (bytes + 255) & ~(size_t)255;
    return o;
  };

  const size_t nSD = (size_t)SLEN * DIM;
  unsigned short* Xb = (unsigned short*)(ws + alloc(nSD * 2));
  size_t wtOff = off;  // Wt region later reused for bf16 head buffers
  unsigned short* Wtq = (unsigned short*)(ws + alloc((size_t)DIM * DIM * 2));
  unsigned short* Wtk = (unsigned short*)(ws + alloc((size_t)DIM * DIM * 2));
  unsigned short* Wtv = (unsigned short*)(ws + alloc((size_t)DIM * DIM * 2));
  float* Yq = (float*)(ws + alloc(nSD * 4));
  float* Yk = (float*)(ws + alloc(nSD * 4));
  float* Yv = (float*)(ws + alloc(nSD * 4));
  float* scores = (float*)(ws + alloc((size_t)SLEN * NE * 4));
  float* gates  = (float*)(ws + alloc((size_t)SLEN * 2 * 4));
  int*   idxb   = (int*)(ws + alloc((size_t)SLEN * 2 * 4));
  float* down   = (float*)(ws + alloc((size_t)SLEN * 96 * 4));
  // Alias bf16 head buffers over the Wt region (Wt dead after the GEMMs).
  unsigned short* Qb = (unsigned short*)(ws + wtOff);
  unsigned short* Kb = Qb + nSD;
  unsigned short* Vb = Kb + nSD;

  // 1. conversions
  cvt_bf16_kernel<<<dim3((unsigned)((nSD + 255) / 256)), 256, 0, stream>>>(x, Xb, nSD);
  dim3 tg(DIM / 32, DIM / 32);
  transpose_cvt_kernel<<<tg, 256, 0, stream>>>(Wq, Wtq);
  transpose_cvt_kernel<<<tg, 256, 0, stream>>>(Wk, Wtk);
  transpose_cvt_kernel<<<tg, 256, 0, stream>>>(Wv, Wtv);

  // 2. gating + LoRA down
  gate_scores_kernel<<<SLEN, 256, 0, stream>>>(x, gW, gb, scores);
  gate_topk_kernel<<<1, 256, 0, stream>>>(scores, gates, idxb);
  lora_down_kernel<<<SLEN, 128, 0, stream>>>(x, qA, kA, vA, down);

  // 3. QKV projections via WMMA
  dim3 gg(DIM / BN, SLEN / BM);
  gemm_bf16_kernel<<<gg, 256, 0, stream>>>(Xb, Wtq, bq, Yq);
  gemm_bf16_kernel<<<gg, 256, 0, stream>>>(Xb, Wtk, bk, Yk);
  gemm_bf16_kernel<<<gg, 256, 0, stream>>>(Xb, Wtv, bv, Yv);

  // 4. LoRA up (adds into Y)
  lora_up_kernel<<<SLEN, 256, 0, stream>>>(down, gates, idxb, qB, kB, vB, Yq, Yk, Yv);

  // 5. head split + bf16 (writes into aliased Wt region — Wt no longer needed)
  split_heads_kernel<<<dim3((unsigned)((nSD + 255) / 256)), 256, 0, stream>>>(Yq, Qb);
  split_heads_kernel<<<dim3((unsigned)((nSD + 255) / 256)), 256, 0, stream>>>(Yk, Kb);
  split_heads_kernel<<<dim3((unsigned)((nSD + 255) / 256)), 256, 0, stream>>>(Yv, Vb);

  // 6. flash attention -> out [S, H*HD]
  dim3 fg(SLEN / 64, NH);
  flash_attn_kernel<<<fg, 128, 0, stream>>>(Qb, Kb, Vb, out);
}
